// HGRN2ForSummarization_7421703488079
// MI455X (gfx1250) — matrix operand order repaired and checked by reading, not compiled, not run on hardware
//
#include <hip/hip_runtime.h>

// ===========================================================================
// HGRN2 encoder-decoder summarization forward for MI455X (gfx1250, CDNA5).
//
// Roofline reasoning:
//   * ~650 GFLOP of dense GEMM; bf16 weights (~130 MB) fit in the 192 MB L2,
//     so steady state is WMMA-compute-bound, not HBM-bound (23.3 TB/s).
//   * Therefore: weights pre-converted fp32->bf16 (transposed [N][K]) once per
//     launch; activations stay fp32 (read once per GEMM) and are converted
//     RNE->bf16 while staging to LDS; accumulation in f32 via
//     v_wmma_f32_16x16x32_bf16 (wave32, 16x16x32 tiles).
//   * GEMM tile: 128x128x32, 256 threads = 8 wave32 waves, each wave a 32x64
//     sub-tile (2x4 WMMA accumulators). Double-buffered LDS, row pad BK+8.
//
// Workspace plan (~560 MB assumed in d_ws): bf16 weights ~132 MB, fp32
// activations ~410 MB (largest: attention scores 4x8x512x2048 f32 = 134 MB),
// bf16 K/V copies 17 MB. Deterministic: everything recomputed every call.
// ===========================================================================

#define D    512
#define NL   6
#define VOC  30522
#define VPAD 30592          // VOC padded up to a multiple of 128
#define NH   8
#define HD   64
#define FF   2048
#define BB   4
#define SRCL 2048
#define TGTL 512
#define NTE  (BB * SRCL)    // 8192 encoder tokens
#define NTD  (BB * TGTL)    // 2048 decoder tokens

#define BM 128
#define BN 128
#define BK 32

typedef __bf16 bf16x16 __attribute__((ext_vector_type(16)));
typedef float  f32x8   __attribute__((ext_vector_type(8)));

union ABFrag { uint4 q[2]; bf16x16 v; };

__device__ __forceinline__ unsigned short f2bf(float f) {
  unsigned int u = __float_as_uint(f);
  u += 0x7fffu + ((u >> 16) & 1u);        // fp32 -> bf16 round-to-nearest-even
  return (unsigned short)(u >> 16);
}
__device__ __forceinline__ float sigm(float x) { return 1.0f / (1.0f + __expf(-x)); }

// ---------------------------------------------------------------------------
// Batched-strided bf16 WMMA GEMM:  C[M,N] = A[M,K](f32) * B(bf16) + bias
//   BT_NK = true : B supplied transposed as Bt[N][K] (contiguous K rows)
//   BT_NK = false: B supplied row-major  as  B[K][N] (LDS-transposed on stage)
// Batch z decomposes as zb = z / bh, zh = z % bh with independent strides.
// K must be a multiple of 32 (true for all GEMMs here: 64, 512, 2048).
// ---------------------------------------------------------------------------
template <bool BT_NK>
__global__ __launch_bounds__(256) void gemm_bf16_kernel(
    const float* __restrict__ A, long long lda,
    const unsigned short* __restrict__ Bm, long long ldb,
    const float* __restrict__ bias,
    float* __restrict__ C, long long ldc,
    int M, int N, int K, int hasBias, int bh,
    long long sA0, long long sA1, long long sB0, long long sB1,
    long long sC0, long long sC1) {
  __shared__ unsigned short As[2][BM][BK + 8];
  __shared__ unsigned short Bs[2][BN][BK + 8];

  const int t = threadIdx.x;
  const int z = blockIdx.z;
  const int zb = z / bh, zh = z - zb * bh;
  A  += (size_t)zb * sA0 + (size_t)zh * sA1;
  Bm += (size_t)zb * sB0 + (size_t)zh * sB1;
  C  += (size_t)zb * sC0 + (size_t)zh * sC1;

  const int m0 = blockIdx.y * BM;
  const int n0 = blockIdx.x * BN;

  const int lane = t & 31;
  const int wv   = t >> 5;
  const int wm   = (wv & 3) * 32;   // 4 waves tile M
  const int wn   = (wv >> 2) * 64;  // 2 waves tile N
  const int rsel = lane & 15;
  const int hsel = lane >> 4;       // half-wave select

  f32x8 acc[2][4];
  for (int i = 0; i < 2; ++i)
    for (int j = 0; j < 4; ++j)
      for (int e = 0; e < 8; ++e) acc[i][j][e] = 0.0f;

  const int nk = K / BK;

  auto stage = [&](int sb, int kk) {
    // A tile: 128x32 f32 -> bf16 (each thread: 4x float4 = 16 values)
    for (int r = 0; r < 4; ++r) {
      const int e   = t + r * 256;       // 1024 float4 slots
      const int row = e >> 3;
      const int c4  = (e & 7) << 2;
      const int gr  = m0 + row;
      float4 f = {0.0f, 0.0f, 0.0f, 0.0f};
      if (gr < M) f = *(const float4*)(A + (size_t)gr * lda + kk + c4);
      unsigned short* dst = &As[sb][row][c4];
      dst[0] = f2bf(f.x); dst[1] = f2bf(f.y);
      dst[2] = f2bf(f.z); dst[3] = f2bf(f.w);
    }
    if (BT_NK) {
      // Bt[N][K]: contiguous 16B chunks (each thread: 2x uint4 = 16 values)
      for (int r = 0; r < 2; ++r) {
        const int e   = t + r * 256;     // 512 uint4 slots
        const int row = e >> 2;
        const int c8  = (e & 3) << 3;
        const int gn  = n0 + row;
        uint4 qv = {0u, 0u, 0u, 0u};
        if (gn < N) qv = *(const uint4*)(Bm + (size_t)gn * ldb + kk + c8);
        *(uint4*)&Bs[sb][row][c8] = qv;
      }
    } else {
      // B[K][N]: coalesced along N, transposed into Bs[n][k]
      for (int r = 0; r < 16; ++r) {
        const int e  = t + r * 256;      // 4096 elements
        const int nn = e & 127;
        const int k2 = e >> 7;
        const int gn = n0 + nn;
        unsigned short v = 0;
        if (gn < N) v = Bm[(size_t)(kk + k2) * ldb + gn];
        Bs[sb][nn][k2] = v;
      }
    }
  };

  stage(0, 0);
  __syncthreads();

  for (int kt = 0; kt < nk; ++kt) {
    const int buf = kt & 1;
    if (kt + 1 < nk) stage(buf ^ 1, (kt + 1) * BK);

    ABFrag af[2], bf_[4];
    for (int i = 0; i < 2; ++i) {
      // ISA 16-bit A layout: lane<16 -> K0..7 / K16..23 ; lane>=16 -> +8
      const unsigned short* ap = &As[buf][wm + i * 16 + rsel][hsel * 8];
      af[i].q[0] = *(const uint4*)(ap);
      af[i].q[1] = *(const uint4*)(ap + 16);
    }
    for (int j = 0; j < 4; ++j) {
      // ISA 16-bit B layout: lane<16 -> K0..7 / K16..23 at col=lane ; lane>=16 -> +8
      const unsigned short* bp = &Bs[buf][wn + j * 16 + rsel][hsel * 8];
      bf_[j].q[0] = *(const uint4*)(bp);
      bf_[j].q[1] = *(const uint4*)(bp + 16);
    }
    for (int i = 0; i < 2; ++i)
      for (int j = 0; j < 4; ++j)
        acc[i][j] = __builtin_amdgcn_wmma_f32_16x16x32_bf16(
            false, af[i].v, false, bf_[j].v, (short)0, acc[i][j], false, false);
    __syncthreads();
  }

  // Epilogue: C/D layout -> VGPR e holds M = e + 8*hsel, N = rsel
  for (int i = 0; i < 2; ++i) {
    const int rbase = m0 + wm + i * 16 + hsel * 8;
    for (int j = 0; j < 4; ++j) {
      const int col = n0 + wn + j * 16 + rsel;
      if (col < N) {
        const float badd = hasBias ? bias[col] : 0.0f;
        for (int e = 0; e < 8; ++e) {
          const int row = rbase + e;
          if (row < M) C[(size_t)row * ldc + col] = acc[i][j][e] + badd;
        }
      }
    }
  }
}

// ---------------------------------------------------------------------------
// Support kernels (all fp32 activations)
// ---------------------------------------------------------------------------
__global__ __launch_bounds__(256) void k_transcvt(const float* __restrict__ W,
                                                  unsigned short* __restrict__ out,
                                                  int K, int N, long long total) {
  const long long idx = (long long)blockIdx.x * 256 + threadIdx.x;
  if (idx >= total) return;
  const int n = (int)(idx / K);
  const int k = (int)(idx - (long long)n * K);
  out[idx] = (n < N) ? f2bf(W[(size_t)k * N + n]) : (unsigned short)0;
}

__global__ __launch_bounds__(256) void k_cvt_bf(const float* __restrict__ in,
                                                unsigned short* __restrict__ out,
                                                long long n) {
  const long long i = (long long)blockIdx.x * 256 + threadIdx.x;
  if (i < n) out[i] = f2bf(in[i]);
}

__global__ __launch_bounds__(256) void k_embed(const float* __restrict__ E,
                                               const int* __restrict__ ids,
                                               float* __restrict__ out, int ntok) {
  const long long i = (long long)blockIdx.x * 256 + threadIdx.x;
  if (i >= (long long)ntok * D) return;
  const int tk = (int)(i >> 9), d = (int)(i & (D - 1));
  out[i] = E[(size_t)ids[tk] * D + d];
}

__global__ __launch_bounds__(256) void k_layernorm(const float* __restrict__ x,
                                                   const float* __restrict__ s,
                                                   const float* __restrict__ b,
                                                   float* __restrict__ y) {
  const int t = threadIdx.x;
  const size_t row = blockIdx.x;
  const float* xr = x + row * D;
  float* yr = y + row * D;
  const float v0 = xr[t], v1 = xr[t + 256];
  __shared__ float rs[256], rq[256];
  rs[t] = v0 + v1;
  rq[t] = v0 * v0 + v1 * v1;
  __syncthreads();
  for (int st = 128; st > 0; st >>= 1) {
    if (t < st) { rs[t] += rs[t + st]; rq[t] += rq[t + st]; }
    __syncthreads();
  }
  const float mean = rs[0] * (1.0f / D);
  const float var  = rq[0] * (1.0f / D) - mean * mean;   // biased var (jnp.var)
  const float inv  = rsqrtf(var + 1e-6f);
  yr[t]       = (v0 - mean) * inv * s[t]       + b[t];
  yr[t + 256] = (v1 - mean) * inv * s[t + 256] + b[t + 256];
}

__global__ __launch_bounds__(256) void k_hgrn_gate(const float* __restrict__ xp,
                                                   float* __restrict__ out, int ntok) {
  const long long i = (long long)blockIdx.x * 256 + threadIdx.x;
  if (i >= (long long)ntok * D) return;
  const int tk = (int)(i >> 9), d = (int)(i & (D - 1));
  const size_t base = (size_t)tk * (3 * D);
  out[i] = xp[base + d] * sigm(xp[base + 2 * D + d]);    // gate_in * sigma(out_gate)
}

__global__ __launch_bounds__(256) void k_glu(float* __restrict__ a,
                                             const float* __restrict__ g, long long n) {
  const long long i = (long long)blockIdx.x * 256 + threadIdx.x;
  if (i < n) a[i] = a[i] * sigm(g[i]);
}

__global__ __launch_bounds__(256) void k_gelu(float* __restrict__ x, long long n) {
  const long long i = (long long)blockIdx.x * 256 + threadIdx.x;
  if (i < n) {
    const float v = x[i];
    x[i] = 0.5f * v * (1.0f + tanhf(0.7978845608028654f * (v + 0.044715f * v * v * v)));
  }
}

__global__ __launch_bounds__(256) void k_add(float* __restrict__ dst,
                                             const float* __restrict__ src, long long n) {
  const long long i = (long long)blockIdx.x * 256 + threadIdx.x;
  if (i < n) dst[i] += src[i];
}

__global__ __launch_bounds__(256) void k_softmax_mask(float* __restrict__ sc,
                                                      const int* __restrict__ mask) {
  const int t = threadIdx.x;
  const int row = blockIdx.x;                 // (b*NH + h)*TGTL + tq
  const int b = row / (NH * TGTL);
  float* p = sc + (size_t)row * SRCL;
  const int* mrow = mask + (size_t)b * SRCL;
  const float scale = 0.125f;                 // 1/sqrt(HD=64)
  __shared__ float red[256];
  float v[8];
  float mx = -3.0e38f;
  for (int j = 0; j < 8; ++j) {
    const int s = t + j * 256;
    float x = p[s] * scale;
    if (mrow[s] == 0) x = -10000.0f;
    v[j] = x;
    mx = fmaxf(mx, x);
  }
  red[t] = mx; __syncthreads();
  for (int st = 128; st > 0; st >>= 1) {
    if (t < st) red[t] = fmaxf(red[t], red[t + st]);
    __syncthreads();
  }
  mx = red[0]; __syncthreads();
  float sum = 0.0f;
  for (int j = 0; j < 8; ++j) { v[j] = __expf(v[j] - mx); sum += v[j]; }
  red[t] = sum; __syncthreads();
  for (int st = 128; st > 0; st >>= 1) {
    if (t < st) red[t] += red[t + st];
    __syncthreads();
  }
  const float inv = 1.0f / red[0];
  for (int j = 0; j < 8; ++j) p[t + j * 256] = v[j] * inv;
}

// ---------------------------------------------------------------------------
// Host-side helpers
// ---------------------------------------------------------------------------
static inline int blks(long long n) { return (int)((n + 255) / 256); }

static void gemm_nk(hipStream_t st, const float* A, long long lda,
                    const unsigned short* Bt, long long ldb, const float* bias,
                    float* C, long long ldc, int M, int N, int K,
                    int batch = 1, int bh = 1,
                    long long sA0 = 0, long long sA1 = 0,
                    long long sB0 = 0, long long sB1 = 0,
                    long long sC0 = 0, long long sC1 = 0) {
  dim3 g((N + BN - 1) / BN, (M + BM - 1) / BM, batch);
  gemm_bf16_kernel<true><<<g, 256, 0, st>>>(A, lda, Bt, ldb, bias, C, ldc, M, N, K,
                                            bias != nullptr, bh, sA0, sA1, sB0, sB1,
                                            sC0, sC1);
}

static void gemm_kn(hipStream_t st, const float* A, long long lda,
                    const unsigned short* Bkn, long long ldb, const float* bias,
                    float* C, long long ldc, int M, int N, int K,
                    int batch = 1, int bh = 1,
                    long long sA0 = 0, long long sA1 = 0,
                    long long sB0 = 0, long long sB1 = 0,
                    long long sC0 = 0, long long sC1 = 0) {
  dim3 g((N + BN - 1) / BN, (M + BM - 1) / BM, batch);
  gemm_bf16_kernel<false><<<g, 256, 0, st>>>(A, lda, Bkn, ldb, bias, C, ldc, M, N, K,
                                             bias != nullptr, bh, sA0, sA1, sB0, sB1,
                                             sC0, sC1);
}

extern "C" void kernel_launch(void* const* d_in, const int* in_sizes, int n_in,
                              void* d_out, int out_size, void* d_ws, size_t ws_size,
                              hipStream_t stream) {
  (void)in_sizes; (void)n_in; (void)out_size; (void)ws_size;
  auto F = [&](int i) { return (const float*)d_in[i]; };
  auto I = [&](int i) { return (const int*)d_in[i]; };

  // setup_inputs() dict order, params flattened in insertion order
  const float* embed   = F(0);
  const float* e_ln1_s = F(1);  const float* e_ln1_b = F(2);
  const float* e_Wp    = F(3);  const float* e_bp    = F(4);
  const float* e_Wo    = F(5);  const float* e_bo    = F(6);
  const float* e_ln2_s = F(7);  const float* e_ln2_b = F(8);
  const float* e_W1    = F(9);  const float* e_b1    = F(10);
  const float* e_Wg    = F(11); const float* e_bg    = F(12);
  const float* e_W2    = F(13); const float* e_b2    = F(14);
  const float* e_lnf_s = F(15); const float* e_lnf_b = F(16);
  const float* g_ln1_s = F(17); const float* g_ln1_b = F(18);
  const float* g_Wp    = F(19); const float* g_bp    = F(20);
  const float* g_Wo    = F(21); const float* g_bo    = F(22);
  const float* g_ln2_s = F(23); const float* g_ln2_b = F(24);
  const float* g_Wq    = F(25); const float* g_bq    = F(26);
  const float* g_Wk    = F(27); const float* g_bk    = F(28);
  const float* g_Wv    = F(29); const float* g_bv    = F(30);
  const float* g_Wc    = F(31); const float* g_bc    = F(32);
  const float* g_ln3_s = F(33); const float* g_ln3_b = F(34);
  const float* g_W1    = F(35); const float* g_b1    = F(36);
  const float* g_W2    = F(37); const float* g_b2    = F(38);
  const float* g_lnf_s = F(39); const float* g_lnf_b = F(40);
  const float* W_out   = F(41); const float* b_out   = F(42);
  const int* in_ids    = I(43);
  const int* amask     = I(44);
  const int* de_ids    = I(45);

  // ---- bump allocator over d_ws ----
  size_t cur = 0;
  auto ab = [&](size_t bytes) -> void* {
    void* p = (char*)d_ws + cur;
    cur += (bytes + 255) & ~(size_t)255;
    return p;
  };
  auto af32 = [&](size_t n) { return (float*)ab(n * 4); };
  auto abf  = [&](size_t n) { return (unsigned short*)ab(n * 2); };

  // ---- weight prep: fp32 [K][N] -> bf16 [Npad][K] (transposed) ----
  auto prepW = [&](const float* W, int K, int N, int Npad) {
    unsigned short* o = abf((size_t)Npad * K);
    const long long total = (long long)Npad * K;
    k_transcvt<<<blks(total), 256, 0, stream>>>(W, o, K, N, total);
    return o;
  };

  unsigned short *eWpT[NL], *eWoT[NL], *eW1T[NL], *eWgT[NL], *eW2T[NL];
  unsigned short *dWpT[NL], *dWoT[NL], *dWqT[NL], *dWkT[NL], *dWvT[NL], *dWcT[NL],
                 *dW1T[NL], *dW2T[NL];
  for (int i = 0; i < NL; ++i) {
    eWpT[i] = prepW(e_Wp + (size_t)i * D * 3 * D, D, 3 * D, 3 * D);
    eWoT[i] = prepW(e_Wo + (size_t)i * D * D,     D, D,     D);
    eW1T[i] = prepW(e_W1 + (size_t)i * D * FF,    D, FF,    FF);
    eWgT[i] = prepW(e_Wg + (size_t)i * D * FF,    D, FF,    FF);
    eW2T[i] = prepW(e_W2 + (size_t)i * FF * D,    FF, D,    D);
    dWpT[i] = prepW(g_Wp + (size_t)i * D * 3 * D, D, 3 * D, 3 * D);
    dWoT[i] = prepW(g_Wo + (size_t)i * D * D,     D, D,     D);
    dWqT[i] = prepW(g_Wq + (size_t)i * D * D,     D, D,     D);
    dWkT[i] = prepW(g_Wk + (size_t)i * D * D,     D, D,     D);
    dWvT[i] = prepW(g_Wv + (size_t)i * D * D,     D, D,     D);
    dWcT[i] = prepW(g_Wc + (size_t)i * D * D,     D, D,     D);
    dW1T[i] = prepW(g_W1 + (size_t)i * D * FF,    D, FF,    FF);
    dW2T[i] = prepW(g_W2 + (size_t)i * FF * D,    FF, D,    D);
  }
  unsigned short* WoutT = prepW(W_out, D, VOC, VPAD);

  // ---- activation buffers ----
  float* x_e   = af32((size_t)NTE * D);
  float* h_e   = af32((size_t)NTE * D);
  float* t1_e  = af32((size_t)NTE * FF);   // also holds xp [NTE,3D]
  float* t2_e  = af32((size_t)NTE * FF);
  float* de_   = af32((size_t)NTE * D);
  float* enc_f = af32((size_t)NTE * D);
  float* kf    = af32((size_t)NTE * D);
  float* vf    = af32((size_t)NTE * D);
  unsigned short* kb = abf((size_t)NTE * D);
  unsigned short* vb = abf((size_t)NTE * D);
  float* y_d   = af32((size_t)NTD * D);
  float* h_d   = af32((size_t)NTD * D);
  float* t1_d  = af32((size_t)NTD * FF);   // also holds xp [NTD,3D]
  float* dd    = af32((size_t)NTD * D);
  float* qb    = af32((size_t)NTD * D);
  float* ctxb  = af32((size_t)NTD * D);
  float* scor  = af32((size_t)BB * NH * TGTL * SRCL);

  // ================= encoder =================
  k_embed<<<blks((long long)NTE * D), 256, 0, stream>>>(embed, in_ids, x_e, NTE);
  for (int i = 0; i < NL; ++i) {
    k_layernorm<<<NTE, 256, 0, stream>>>(x_e, e_ln1_s + i * D, e_ln1_b + i * D, h_e);
    gemm_nk(stream, h_e, D, eWpT[i], D, e_bp + (size_t)i * 3 * D, t1_e, 3 * D,
            NTE, 3 * D, D);
    k_hgrn_gate<<<blks((long long)NTE * D), 256, 0, stream>>>(t1_e, h_e, NTE);
    gemm_nk(stream, h_e, D, eWoT[i], D, e_bo + (size_t)i * D, de_, D, NTE, D, D);
    k_add<<<blks((long long)NTE * D), 256, 0, stream>>>(x_e, de_, (long long)NTE * D);

    k_layernorm<<<NTE, 256, 0, stream>>>(x_e, e_ln2_s + i * D, e_ln2_b + i * D, h_e);
    gemm_nk(stream, h_e, D, eW1T[i], D, e_b1 + (size_t)i * FF, t1_e, FF, NTE, FF, D);
    gemm_nk(stream, x_e, D, eWgT[i], D, e_bg + (size_t)i * FF, t2_e, FF, NTE, FF, D);
    k_glu<<<blks((long long)NTE * FF), 256, 0, stream>>>(t1_e, t2_e, (long long)NTE * FF);
    gemm_nk(stream, t1_e, FF, eW2T[i], FF, e_b2 + (size_t)i * D, de_, D, NTE, D, FF);
    k_add<<<blks((long long)NTE * D), 256, 0, stream>>>(x_e, de_, (long long)NTE * D);
  }
  k_layernorm<<<NTE, 256, 0, stream>>>(x_e, e_lnf_s, e_lnf_b, enc_f);

  // ================= decoder =================
  k_embed<<<blks((long long)NTD * D), 256, 0, stream>>>(embed, de_ids, y_d, NTD);
  for (int i = 0; i < NL; ++i) {
    k_layernorm<<<NTD, 256, 0, stream>>>(y_d, g_ln1_s + i * D, g_ln1_b + i * D, h_d);
    gemm_nk(stream, h_d, D, dWpT[i], D, g_bp + (size_t)i * 3 * D, t1_d, 3 * D,
            NTD, 3 * D, D);
    k_hgrn_gate<<<blks((long long)NTD * D), 256, 0, stream>>>(t1_d, h_d, NTD);
    gemm_nk(stream, h_d, D, dWoT[i], D, g_bo + (size_t)i * D, dd, D, NTD, D, D);
    k_add<<<blks((long long)NTD * D), 256, 0, stream>>>(y_d, dd, (long long)NTD * D);

    // ---- cross attention ----
    k_layernorm<<<NTD, 256, 0, stream>>>(y_d, g_ln2_s + i * D, g_ln2_b + i * D, h_d);
    gemm_nk(stream, h_d, D, dWqT[i], D, g_bq + (size_t)i * D, qb, D, NTD, D, D);
    gemm_nk(stream, enc_f, D, dWkT[i], D, g_bk + (size_t)i * D, kf, D, NTE, D, D);
    gemm_nk(stream, enc_f, D, dWvT[i], D, g_bv + (size_t)i * D, vf, D, NTE, D, D);
    k_cvt_bf<<<blks((long long)NTE * D), 256, 0, stream>>>(kf, kb, (long long)NTE * D);
    k_cvt_bf<<<blks((long long)NTE * D), 256, 0, stream>>>(vf, vb, (long long)NTE * D);

    // scores[b,h,t,s] = q[b,t,h,:] . k[b,s,h,:]   (Bt = K-activations [S][HD])
    gemm_nk(stream, qb, D, kb, D, nullptr, scor, SRCL, TGTL, SRCL, HD,
            BB * NH, NH,
            (long long)TGTL * D, HD,
            (long long)SRCL * D, HD,
            (long long)NH * TGTL * SRCL, (long long)TGTL * SRCL);
    k_softmax_mask<<<BB * NH * TGTL, 256, 0, stream>>>(scor, amask);
    // ctx[b,t,h,:] = P[b,h,t,:] @ V[b,:,h,:]      (B row-major [S][HD])
    gemm_kn(stream, scor, SRCL, vb, D, nullptr, ctxb, D, TGTL, HD, SRCL,
            BB * NH, NH,
            (long long)NH * TGTL * SRCL, (long long)TGTL * SRCL,
            (long long)SRCL * D, HD,
            (long long)TGTL * D, HD);
    gemm_nk(stream, ctxb, D, dWcT[i], D, g_bc + (size_t)i * D, dd, D, NTD, D, D);
    k_add<<<blks((long long)NTD * D), 256, 0, stream>>>(y_d, dd, (long long)NTD * D);

    // ---- FFN (gelu) ----
    k_layernorm<<<NTD, 256, 0, stream>>>(y_d, g_ln3_s + i * D, g_ln3_b + i * D, h_d);
    gemm_nk(stream, h_d, D, dW1T[i], D, g_b1 + (size_t)i * FF, t1_d, FF, NTD, FF, D);
    k_gelu<<<blks((long long)NTD * FF), 256, 0, stream>>>(t1_d, (long long)NTD * FF);
    gemm_nk(stream, t1_d, FF, dW2T[i], FF, g_b2 + (size_t)i * D, dd, D, NTD, D, FF);
    k_add<<<blks((long long)NTD * D), 256, 0, stream>>>(y_d, dd, (long long)NTD * D);
  }
  k_layernorm<<<NTD, 256, 0, stream>>>(y_d, g_lnf_s, g_lnf_b, h_d);

  // ---- final projection to vocab ----
  gemm_nk(stream, h_d, D, WoutT, D, b_out, (float*)d_out, VOC, NTD, VOC, D);
}
